// GraphAttentionLayer_22308060136378
// MI455X (gfx1250) — matrix operand order repaired and checked by reading, not compiled, not run on hardware
//
#include <hip/hip_runtime.h>
#include <cstdint>
#include <cstddef>

#define B_DIM 32
#define N_DIM 1024
#define IN_F  256
#define OUT_F 256
#define GAT_ALPHA 0.2f
#define GAT_NEG_INF -9.0e15f

typedef __attribute__((ext_vector_type(16))) __bf16        v16bf;
typedef __attribute__((ext_vector_type(8)))  float         v8f;
typedef __attribute__((ext_vector_type(4)))  float         f4;
typedef __attribute__((ext_vector_type(4)))  int           i4;
typedef __attribute__((ext_vector_type(4)))  unsigned short u16x4;
typedef __attribute__((ext_vector_type(8)))  unsigned short u16x8;
typedef __attribute__((ext_vector_type(16))) unsigned short u16x16;

__device__ __forceinline__ unsigned short f2bf_bits(float f) {
  return (unsigned short)(__builtin_bit_cast(unsigned int, f) >> 16);
}
__device__ __forceinline__ float bf2f(unsigned short u) {
  return __builtin_bit_cast(float, ((unsigned int)u) << 16);
}
__device__ __forceinline__ v16bf cast16(u16x16 u) {
  return __builtin_bit_cast(v16bf, u);
}
__device__ __forceinline__ v16bf join8(u16x8 lo, u16x8 hi) {
  u16x16 u = __builtin_shufflevector(lo, hi, 0, 1, 2, 3, 4, 5, 6, 7,
                                     8, 9, 10, 11, 12, 13, 14, 15);
  return __builtin_bit_cast(v16bf, u);
}

// ---------------------------------------------------------------------------
// K1: Wh = h @ W^T  (M = B*N = 32768, K = 256, N = 256), bf16 WMMA, f32 acc.
// Stores Wh TRANSPOSED per batch: WhT[b][o][m] (bf16) so the attention GEMM's
// B operand fragments are contiguous 16-byte loads.
// Block = 16 rows x 256 cols; 8 waves, each owns col-tiles {w, w+8}.
// ---------------------------------------------------------------------------
__global__ __launch_bounds__(256) void k_wh_gemm(const float* __restrict__ h,
                                                 const float* __restrict__ W,
                                                 unsigned short* __restrict__ WhT) {
  const int lane = threadIdx.x & 31;
  const int wv   = threadIdx.x >> 5;
  const int m0   = blockIdx.x * 16;          // global row over B*N
  const int lm   = lane & 15;
  const int half = lane >> 4;

  const int col0 = wv * 16;
  const int col1 = col0 + 128;
  const int ks   = half * 8;    // A-fragment K sub-offset (16-bit A 16x32 layout)
  const int hk   = half * 16;   // unused here; B fragments use ks, same k order as A

  const float* hrow  = h + (size_t)(m0 + lm) * IN_F;
  const float* wrow0 = W + (size_t)(col0 + lm) * IN_F;
  const float* wrow1 = W + (size_t)(col1 + lm) * IN_F;

  v8f acc0 = {};
  v8f acc1 = {};

  for (int kk = 0; kk < IN_F; kk += 32) {
    // A fragment: lane holds row m, K = {ks..ks+7} U {16+ks..16+ks+7}
    f4 a0 = *(const f4*)(hrow + kk + ks);
    f4 a1 = *(const f4*)(hrow + kk + ks + 4);
    f4 a2 = *(const f4*)(hrow + kk + 16 + ks);
    f4 a3 = *(const f4*)(hrow + kk + 16 + ks + 4);
    u16x16 au;
#pragma unroll
    for (int i = 0; i < 4; ++i) {
      au[i]      = f2bf_bits(a0[i]);
      au[4 + i]  = f2bf_bits(a1[i]);
      au[8 + i]  = f2bf_bits(a2[i]);
      au[12 + i] = f2bf_bits(a3[i]);
    }
    v16bf afrag = cast16(au);

    // B fragments: lane holds col o, K = {ks..ks+7} U {16+ks..16+ks+7} (matches A)
    u16x16 bu0, bu1;
#pragma unroll
    for (int i = 0; i < 4; ++i) {
      f4 w0 = *(const f4*)(wrow0 + kk + ks + 4 * i + (i & 2) * 4);
      f4 w1 = *(const f4*)(wrow1 + kk + ks + 4 * i + (i & 2) * 4);
#pragma unroll
      for (int jj = 0; jj < 4; ++jj) {
        bu0[4 * i + jj] = f2bf_bits(w0[jj]);
        bu1[4 * i + jj] = f2bf_bits(w1[jj]);
      }
    }
    acc0 = __builtin_amdgcn_wmma_f32_16x16x32_bf16(false, afrag, false, cast16(bu0),
                                                   (short)0, acc0, false, false);
    acc1 = __builtin_amdgcn_wmma_f32_16x16x32_bf16(false, afrag, false, cast16(bu1),
                                                   (short)0, acc1, false, false);
  }

  // C layout: lanes0-15 -> M = reg r, N = lane; lanes16-31 -> M = 8+r.
  // WhT[b][o][m]: 8 rows per lane are contiguous -> one b128 store per tile.
  const int b    = m0 >> 10;
  const int mm   = m0 & 1023;
  const int mOff = half * 8;
  u16x8 s0, s1;
#pragma unroll
  for (int r = 0; r < 8; ++r) {
    s0[r] = f2bf_bits(acc0[r]);
    s1[r] = f2bf_bits(acc1[r]);
  }
  *(u16x8*)(WhT + ((size_t)b * OUT_F + col0 + lm) * N_DIM + mm + mOff) = s0;
  *(u16x8*)(WhT + ((size_t)b * OUT_F + col1 + lm) * N_DIM + mm + mOff) = s1;
}

// ---------------------------------------------------------------------------
// K2: si[m] = Wh[m,:] . a1 ; sj[m] = Wh[m,:] . a2 from transposed bf16 Wh.
// Thread t handles one m; reads are coalesced (consecutive m per o).
// ---------------------------------------------------------------------------
__global__ __launch_bounds__(256) void k_scores(const unsigned short* __restrict__ WhT,
                                                const float* __restrict__ a,
                                                float* __restrict__ si,
                                                float* __restrict__ sj) {
  const int gm = blockIdx.x * 256 + threadIdx.x;
  const int b  = gm >> 10;
  const int mm = gm & 1023;
  const unsigned short* base = WhT + (size_t)b * OUT_F * N_DIM + mm;
  float s1 = 0.f, s2 = 0.f;
#pragma unroll 4
  for (int o = 0; o < OUT_F; ++o) {
    float wv = bf2f(base[(size_t)o * N_DIM]);
    s1 = fmaf(wv, a[o], s1);
    s2 = fmaf(wv, a[OUT_F + o], s2);
  }
  si[gm] = s1;
  sj[gm] = s2;
}

// ---------------------------------------------------------------------------
// K3: fused masked-softmax attention + P@Wh + ELU. One block per (b, 16-row
// tile). adj (128 MB, the roofline-dominant stream) is read exactly once,
// vectorized b128 and non-temporal so it never evicts the L2-resident WhT.
// e/p kept as bf16 in 32KB LDS.
// ---------------------------------------------------------------------------
__global__ __launch_bounds__(256) void k_attn(const int* __restrict__ adj,
                                              const unsigned short* __restrict__ WhT,
                                              const float* __restrict__ si,
                                              const float* __restrict__ sj,
                                              float* __restrict__ out) {
  __shared__ unsigned short ep[16 * N_DIM];   // 32 KB: e (bf16), then p in place
  __shared__ float red_m[16][16];
  __shared__ float red_l[16][16];
  __shared__ float mRow[16];
  __shared__ float lInv[16];

  const int t  = threadIdx.x;
  const int b  = blockIdx.x >> 6;
  const int i0 = (blockIdx.x & 63) << 4;

  // ---- pass 1: masked leaky-relu logits + online (max, sum); 16 thr / row,
  //      4 consecutive j per thread per iteration (b128 adj + b128 sj loads)
  {
    const int r = t >> 4;
    const int c = (t & 15) * 4;
    const int*   arow = adj + ((size_t)b * N_DIM + i0 + r) * N_DIM;
    const float* sjb  = sj + (size_t)b * N_DIM;
    const float  sir  = si[(size_t)b * N_DIM + i0 + r];
    float mc = -INFINITY, lc = 0.f;
    for (int jj = 0; jj < N_DIM; jj += 64) {
      const int j = jj + c;
      i4 av = __builtin_nontemporal_load((const i4*)(arow + j));  // streamed once
      f4 sv = *(const f4*)(sjb + j);
      float e[4];
      u16x4 w;
#pragma unroll
      for (int q = 0; q < 4; ++q) {
        float x = sir + sv[q];
        x = x > 0.f ? x : GAT_ALPHA * x;
        if (av[q] <= 0) x = GAT_NEG_INF;   // all-masked row -> uniform softmax, as ref
        e[q] = x;
        w[q] = f2bf_bits(x);
      }
      *(u16x4*)&ep[r * N_DIM + j] = w;
      float m4 = fmaxf(fmaxf(e[0], e[1]), fmaxf(e[2], e[3]));
      float mn = fmaxf(mc, m4);
      float s  = __expf(e[0] - mn) + __expf(e[1] - mn) +
                 __expf(e[2] - mn) + __expf(e[3] - mn);
      lc = lc * __expf(mc - mn) + s;
      mc = mn;
    }
    red_m[r][t & 15] = mc;
    red_l[r][t & 15] = lc;
  }
  __syncthreads();
  if (t < 16) {
    float m = -INFINITY, l = 0.f;
#pragma unroll
    for (int c = 0; c < 16; ++c) {
      float mm = red_m[t][c], ll = red_l[t][c];
      float mn = fmaxf(m, mm);
      l = l * __expf(m - mn) + ll * __expf(mm - mn);
      m = mn;
    }
    mRow[t] = m;
    lInv[t] = 1.0f / l;
  }
  __syncthreads();
  // ---- convert e -> normalized p (bf16) in place; each thread owns its slots
  {
    const int r = t >> 4;
    const int c = (t & 15) * 4;
    const float mr = mRow[r], li = lInv[r];
    for (int jj = 0; jj < N_DIM; jj += 64) {
      const int j = jj + c;
      u16x4 w = *(u16x4*)&ep[r * N_DIM + j];
#pragma unroll
      for (int q = 0; q < 4; ++q)
        w[q] = f2bf_bits(__expf(bf2f(w[q]) - mr) * li);
      *(u16x4*)&ep[r * N_DIM + j] = w;
    }
  }
  __syncthreads();

  // ---- pass 2: (16 x 1024) P @ (1024 x 256) Wh with bf16 WMMA, 32 K-chunks
  const int lane = t & 31;
  const int wv   = t >> 5;
  const int lm   = lane & 15;
  const int half = lane >> 4;
  const int col0 = wv * 16, col1 = col0 + 128;
  const int ks   = half * 8;   // unused here; A fragment also uses hk (k = 16h + s)
  const int hk   = half * 16;  // B-fragment K sub-offset

  const unsigned short* w0base = WhT + ((size_t)b * OUT_F + col0 + lm) * N_DIM + hk;
  const unsigned short* w1base = WhT + ((size_t)b * OUT_F + col1 + lm) * N_DIM + hk;

  v8f acc0 = {};
  v8f acc1 = {};
  for (int j0 = 0; j0 < N_DIM; j0 += 32) {
    u16x8 alo = *(const u16x8*)&ep[lm * N_DIM + j0 + hk];
    u16x8 ahi = *(const u16x8*)&ep[lm * N_DIM + j0 + hk + 8];
    v16bf afrag = join8(alo, ahi);

    u16x8 b0lo = *(const u16x8*)(w0base + j0);
    u16x8 b0hi = *(const u16x8*)(w0base + j0 + 8);
    u16x8 b1lo = *(const u16x8*)(w1base + j0);
    u16x8 b1hi = *(const u16x8*)(w1base + j0 + 8);
    v16bf bf0 = join8(b0lo, b0hi);
    v16bf bf1 = join8(b1lo, b1hi);

    acc0 = __builtin_amdgcn_wmma_f32_16x16x32_bf16(false, afrag, false, bf0,
                                                   (short)0, acc0, false, false);
    acc1 = __builtin_amdgcn_wmma_f32_16x16x32_bf16(false, afrag, false, bf1,
                                                   (short)0, acc1, false, false);
  }

  // ---- epilogue: ELU (alpha = 1) + non-temporal f32 store (write-once output)
  const int mOff = half * 8;
  float* orow = out + ((size_t)b * N_DIM + i0 + mOff) * OUT_F;
#pragma unroll
  for (int r = 0; r < 8; ++r) {
    float x0 = acc0[r];
    float x1 = acc1[r];
    float y0 = x0 > 0.f ? x0 : (__expf(x0) - 1.f);
    float y1 = x1 > 0.f ? x1 : (__expf(x1) - 1.f);
    __builtin_nontemporal_store(y0, orow + (size_t)r * OUT_F + col0 + lm);
    __builtin_nontemporal_store(y1, orow + (size_t)r * OUT_F + col1 + lm);
  }
}

// ---------------------------------------------------------------------------
extern "C" void kernel_launch(void* const* d_in, const int* in_sizes, int n_in,
                              void* d_out, int out_size, void* d_ws, size_t ws_size,
                              hipStream_t stream) {
  (void)in_sizes; (void)n_in; (void)out_size; (void)ws_size;
  const float* h   = (const float*)d_in[0];
  const int*   adj = (const int*)d_in[1];
  const float* W   = (const float*)d_in[2];
  const float* a   = (const float*)d_in[3];
  float* out = (float*)d_out;

  // workspace layout: WhT bf16 [B][OUT_F][N] (16 MB) | si f32 | sj f32
  unsigned short* WhT = (unsigned short*)d_ws;
  const size_t whtElems = (size_t)B_DIM * OUT_F * N_DIM;
  float* si = (float*)((char*)d_ws + whtElems * sizeof(unsigned short));
  float* sj = si + (size_t)B_DIM * N_DIM;

  k_wh_gemm<<<(B_DIM * N_DIM) / 16, 256, 0, stream>>>(h, W, WhT);
  k_scores<<<(B_DIM * N_DIM) / 256, 256, 0, stream>>>(WhT, a, si, sj);
  k_attn<<<B_DIM * (N_DIM / 16), 256, 0, stream>>>(adj, WhT, si, sj, out);
}
